// MolecularE3nnEgcn_32186484916936
// MI455X (gfx1250) — compile-verified, matrix operand order checked by reading, not run on hardware
//
#include <hip/hip_runtime.h>
#include <hip/hip_bf16.h>
#include <math.h>

// ---------------------------------------------------------------------------
// Problem constants (match reference)
// ---------------------------------------------------------------------------
#define N_NODES 50000
#define N_EDGES 150000
#define HDIM    32
#define L_CONV  2
#define N_GRAPH 512
#define NB      10
#define FC_HID  256
#define ET      ((N_EDGES + 15) / 16)   // 9375 edge tiles of 16

typedef __attribute__((ext_vector_type(16))) _Float16 v16h;
typedef __attribute__((ext_vector_type(8)))  float    v8f;
typedef __attribute__((ext_vector_type(4)))  float    v4f;
typedef __attribute__((ext_vector_type(4)))  int      v4i;

// CDNA5 async global->LDS path (ASYNCcnt), gated so compile never breaks.
// ROCm clang signature (from diagnostics): (int4 AS1*, int4 AS3*, imm, imm)
#if defined(__AMDGCN__) && __has_builtin(__builtin_amdgcn_global_load_async_to_lds_b128) && __has_builtin(__builtin_amdgcn_s_wait_asynccnt)
#define USE_ASYNC_LDS 1
typedef __attribute__((address_space(1))) v4i gv4i;
typedef __attribute__((address_space(3))) v4i lv4i;
#else
#define USE_ASYNC_LDS 0
#endif

// Scale factors folded into data:
//   fc1  *= 1/sqrt(10); emb *= sqrt(10); fc2 *= 1/16 (=1/sqrt(256));
//   xs   *= 1/sqrt(32) (1/sqrt(H)); scatter *= 1/sqrt(3) (=1/deg_norm)
#define INV_SQRT10   0.316227766016838f
#define SQRT10       3.162277660168379f
#define INV_16       0.0625f
#define INV_SQRTH    0.176776695296637f
#define SQRT2        1.414213562373095f
#define INV_DEGNORM  0.577350269189626f

// ---------------------------------------------------------------------------
// Small prep kernels
// ---------------------------------------------------------------------------
__global__ void k_zero(float* __restrict__ p, int n) {
    int i = blockIdx.x * blockDim.x + threadIdx.x;
    if (i < n) p[i] = 0.0f;
}

__global__ void k_gather_x(const int* __restrict__ xt, const float* __restrict__ embd,
                           float* __restrict__ x) {
    int i = blockIdx.x * blockDim.x + threadIdx.x;
    if (i < N_NODES * HDIM) {
        int n = i >> 5, c = i & 31;
        x[i] = embd[xt[n] * HDIM + c];
    }
}

__global__ void k_out_init(const float* __restrict__ wpb, float* __restrict__ out) {
    int g = blockIdx.x * blockDim.x + threadIdx.x;
    if (g < N_GRAPH) out[g] = wpb[0];
}

// fc1 -> WMMA B-fragment order: [L][16 ntiles][32 lanes][16 halves]
__global__ void k_fc1_frag(const float* __restrict__ fc1, _Float16* __restrict__ dst) {
    int idx = blockIdx.x * blockDim.x + threadIdx.x;
    if (idx >= L_CONV * 16 * 32 * 16) return;
    int i    = idx & 15;
    int lane = (idx >> 4) & 31;
    int nt   = (idx >> 9) & 15;
    int l    = idx >> 13;
    int k = ((i < 8) ? i : (i + 8)) + ((lane >> 4) << 3);
    int c = nt * 16 + (lane & 15);
    float v = (k < NB) ? fc1[(l * NB + k) * FC_HID + c] * INV_SQRT10 : 0.0f;
    dst[idx] = (_Float16)v;
}

// fc2 -> B-fragment order: [L][64 ntiles][8 kk][32 lanes][16 halves]
// ntile nt = u*2 + nb : column = u*32 + nb*16 + (lane&15)
__global__ void k_fc2_frag(const float* __restrict__ fc2, _Float16* __restrict__ dst) {
    int idx = blockIdx.x * blockDim.x + threadIdx.x;
    if (idx >= L_CONV * 64 * 8 * 32 * 16) return;
    int i    = idx & 15;
    int lane = (idx >> 4) & 31;
    int kk   = (idx >> 9) & 7;
    int nt   = (idx >> 12) & 63;
    int l    = idx >> 18;
    int k    = kk * 32 + ((i < 8) ? i : (i + 8)) + ((lane >> 4) << 3);
    int cOut = (nt >> 1) * 32 + (nt & 1) * 16 + (lane & 15);
    float v = fc2[((size_t)l * FC_HID + k) * 1024 + cOut] * INV_16;
    dst[idx] = (_Float16)v;
}

// Radial basis, pre-swizzled into A-fragment order: [ET][32 lanes][16 halves]
__global__ void k_emb_frag(const float* __restrict__ ea, _Float16* __restrict__ dst) {
    int idx = blockIdx.x * blockDim.x + threadIdx.x;     // tile*32 + lane
    if (idx >= ET * 32) return;
    int lane = idx & 31;
    int tile = idx >> 5;
    int e = tile * 16 + (lane & 15);
    float len = 0.0f;
    if (e < N_EDGES) {
        float a = ea[3 * e], b = ea[3 * e + 1], c = ea[3 * e + 2];
        len = sqrtf(a * a + b * b + c * c);
    }
    union { v16h v; _Float16 h[16]; } frag;
    #pragma unroll
    for (int i = 0; i < 16; ++i) {
        int k = ((i < 8) ? i : (i + 8)) + ((lane >> 4) << 3);
        float v = 0.0f;
        if (e < N_EDGES && k < NB) {
            float center = 2.0f * (float)(k + 1) / 11.0f;
            float diff = (len - center) * (11.0f / 2.0f);
            float d2 = diff * diff;
            if (d2 < 1.0f)
                v = 1.14136f * __expf(2.0f - 2.0f / (1.0f - d2)) * SQRT10;
        }
        frag.h[i] = (_Float16)v;
    }
    *(v16h*)(dst + (size_t)idx * 16) = frag.v;
}

// ---------------------------------------------------------------------------
// Fused convolution layer (cooperative).
//   Block = 128 threads (4 waves).  Each wave owns T=2 tiles of 16 edges.
//   Phase 1: h = relu(emb @ fc1)*sqrt2 -> LDS  (WMMA GEMM1, per wave)
//   Phase 2: for each u chunk (16 KB of fc2 B-fragments = columns u*32..u*32+31):
//            async-copy chunk u+1 to LDS (double buffer) while all 4 waves run
//            8 K-step WMMA GEMMs per tile from LDS, fusing the per-edge
//            contraction ef += x[src]_u * acc.
//   Phase 3: atomic scatter-add of ef into xout[dst].
// ---------------------------------------------------------------------------
__launch_bounds__(128)
__global__ void k_conv(const float* __restrict__ xin, float* __restrict__ xout,
                       const int* __restrict__ eidx,
                       const _Float16* __restrict__ embf,
                       const _Float16* __restrict__ fc1f,
                       const _Float16* __restrict__ fc2f) {
    constexpr int T = 2;
    constexpr int NW = 4;                                    // waves per block
    __shared__ __align__(16) _Float16 hld[NW][T][16][FC_HID]; // 64 KB
    __shared__ __align__(16) _Float16 bstage[2][8192];        // 32 KB double buffer
    __shared__ float xsld[NW][T][16][HDIM];                   // 16 KB
    __shared__ int   dstld[NW][T][16];

    const int tid   = threadIdx.x;       // 0..127
    const int lane  = tid & 31;          // wave32 lane
    const int wave  = tid >> 5;
    const int hig   = lane >> 4;         // K-group (ISA A/B layout)
    const int lan15 = lane & 15;

    // -------- Phase 1: GEMM1 (emb @ fc1) -> LDS; gather xs, dst
    for (int t = 0; t < T; ++t) {
        int tile = blockIdx.x * (NW * T) + wave * T + t;
        v16h aemb;
        if (tile < ET) {
            aemb = *(const v16h*)(embf + ((size_t)tile * 32 + lane) * 16);
        } else {
            #pragma unroll
            for (int i = 0; i < 16; ++i) aemb[i] = (_Float16)0.0f;
        }

        #pragma unroll
        for (int nt = 0; nt < 16; ++nt) {
            v16h b = *(const v16h*)(fc1f + ((size_t)nt * 32 + lane) * 16);
            v8f acc = {};
            acc = __builtin_amdgcn_wmma_f32_16x16x32_f16(
                false, aemb, false, b, (short)0, acc, false, false);
            #pragma unroll
            for (int r = 0; r < 8; ++r) {
                int m = r + (hig << 3);
                float v = acc[r];
                v = (v > 0.0f) ? v * SQRT2 : 0.0f;
                hld[wave][t][m][nt * 16 + lan15] = (_Float16)v;
            }
        }

        for (int j = 0; j < 16; ++j) {
            int e = tile * 16 + j;
            float v = 0.0f;
            if (e < N_EDGES) {
                int s = eidx[e];                              // src
                v = xin[(size_t)s * HDIM + lane] * INV_SQRTH;
            }
            xsld[wave][t][j][lane] = v;
        }
        if (lane < 16) {
            int e = tile * 16 + lane;
            dstld[wave][t][lane] = (e < N_EDGES) ? eidx[N_EDGES + e] : -1;
        }
    }
    __syncthreads();

    // -------- Phase 2 helpers --------------------------------------------
    // One u-chunk = 8192 halves (16 KB) = two nb-blocks of [8 kk][32 lanes][16].
#if USE_ASYNC_LDS
    #define COPY_CHUNK(U, BI)                                                  \
    {   const _Float16* _src = fc2f + (size_t)(U) * 8192;                      \
        _Pragma("unroll")                                                      \
        for (int j = 0; j < 8; ++j) {                                          \
            size_t _off = (size_t)tid * 128 + (size_t)j * 16;                  \
            __builtin_amdgcn_global_load_async_to_lds_b128(                    \
                (gv4i*)((char*)(void*)const_cast<_Float16*>(_src) + _off),     \
                (lv4i*)((char*)&bstage[BI][0] + _off), 0, 0);                  \
        } }
#else
    #define COPY_CHUNK(U, BI)                                                  \
    {   const _Float16* _src = fc2f + (size_t)(U) * 8192;                      \
        _Pragma("unroll")                                                      \
        for (int j = 0; j < 8; ++j) {                                          \
            size_t _off = (size_t)tid * 128 + (size_t)j * 16;                  \
            *(v4f*)((char*)&bstage[BI][0] + _off) =                            \
                *(const v4f*)((const char*)_src + _off);                       \
        } }
#endif

    // One WMMA pass over an nb-block for both tiles of this wave (NBL literal).
    #define GEMM2_STEP(BBUF, NBL)                                              \
    {   v8f acc0 = {}; v8f acc1 = {};                                          \
        _Pragma("unroll")                                                      \
        for (int kk = 0; kk < 8; ++kk) {                                       \
            v16h b = *(const v16h*)((BBUF) + kk * 512 + lane * 16);            \
            union { v16h v; v4f q[2]; } a0, a1;                                \
            a0.q[0] = *(const v4f*)&hld[wave][0][lan15][kk*32 + (hig<<3)];     \
            a0.q[1] = *(const v4f*)&hld[wave][0][lan15][kk*32 + 16 + (hig<<3)];\
            acc0 = __builtin_amdgcn_wmma_f32_16x16x32_f16(                     \
                false, a0.v, false, b, (short)0, acc0, false, false);          \
            a1.q[0] = *(const v4f*)&hld[wave][1][lan15][kk*32 + (hig<<3)];     \
            a1.q[1] = *(const v4f*)&hld[wave][1][lan15][kk*32 + 16 + (hig<<3)];\
            acc1 = __builtin_amdgcn_wmma_f32_16x16x32_f16(                     \
                false, a1.v, false, b, (short)0, acc1, false, false);          \
        }                                                                      \
        _Pragma("unroll")                                                      \
        for (int r = 0; r < 8; ++r) {                                          \
            ef[0][NBL][r] += sx[0][r] * acc0[r];                               \
            ef[1][NBL][r] += sx[1][r] * acc1[r];                               \
        } }

    float ef[T][2][8];
    #pragma unroll
    for (int t = 0; t < T; ++t)
        #pragma unroll
        for (int nb = 0; nb < 2; ++nb)
            #pragma unroll
            for (int r = 0; r < 8; ++r) ef[t][nb][r] = 0.0f;

#if USE_ASYNC_LDS
    COPY_CHUNK(0, 0);                                  // prefetch chunk u=0
    for (int u = 0; u < 32; ++u) {
        const _Float16* bbuf = &bstage[u & 1][0];
        if (u < 31) {
            COPY_CHUNK(u + 1, (u + 1) & 1);            // overlap next chunk
            __builtin_amdgcn_s_wait_asynccnt(8);       // chunk u complete (ours)
        } else {
            __builtin_amdgcn_s_wait_asynccnt(0);
        }
        __syncthreads();                               // chunk u complete (all)

        float sx[T][8];
        #pragma unroll
        for (int t = 0; t < T; ++t)
            #pragma unroll
            for (int r = 0; r < 8; ++r)
                sx[t][r] = xsld[wave][t][r + (hig << 3)][u];

        GEMM2_STEP(bbuf,        0);                    // columns u*32 .. +15
        GEMM2_STEP(bbuf + 4096, 1);                    // columns u*32+16 .. +31
        __syncthreads();                               // done reading buffer
    }
#else
    for (int u = 0; u < 32; ++u) {
        __syncthreads();                               // prior compute done
        COPY_CHUNK(u, 0);
        __syncthreads();                               // chunk visible to all

        float sx[T][8];
        #pragma unroll
        for (int t = 0; t < T; ++t)
            #pragma unroll
            for (int r = 0; r < 8; ++r)
                sx[t][r] = xsld[wave][t][r + (hig << 3)][u];

        const _Float16* bbuf = &bstage[0][0];
        GEMM2_STEP(bbuf,        0);
        GEMM2_STEP(bbuf + 4096, 1);
    }
#endif

    // -------- Phase 3: scatter-add ef to destination nodes
    #pragma unroll
    for (int t = 0; t < T; ++t) {
        #pragma unroll
        for (int r = 0; r < 8; ++r) {
            int m = r + (hig << 3);
            int node = dstld[wave][t][m];
            if (node >= 0) {
                #pragma unroll
                for (int nb = 0; nb < 2; ++nb) {
                    int col = nb * 16 + lan15;
                    atomicAdd(&xout[(size_t)node * HDIM + col],
                              ef[t][nb][r] * INV_DEGNORM);
                }
            }
        }
    }
    #undef COPY_CHUNK
    #undef GEMM2_STEP
}

// ---------------------------------------------------------------------------
// Output head: 2x (x @ lin_w + b, relu), pooled projection per graph.
// ---------------------------------------------------------------------------
__launch_bounds__(256)
__global__ void k_post(const float* __restrict__ x, const int* __restrict__ batch,
                       const float* __restrict__ linw, const float* __restrict__ linb,
                       const float* __restrict__ wpw, float* __restrict__ out) {
    __shared__ float xls[8][32][33];
    __shared__ float wls[2 * 32 * 32];
    __shared__ float bls[64];
    __shared__ float wp[32];

    int lane = threadIdx.x & 31, wave = threadIdx.x >> 5;
    int n0 = blockIdx.x * 256 + wave * 32;

    for (int j = 0; j < 32; ++j) {
        int n = n0 + j;
        xls[wave][j][lane] = (n < N_NODES) ? x[(size_t)n * HDIM + lane] : 0.0f;
    }
    for (int i = threadIdx.x; i < 2048; i += 256) wls[i] = linw[i];
    if (threadIdx.x < 64) bls[threadIdx.x] = linb[threadIdx.x];
    if (threadIdx.x < 32) wp[threadIdx.x] = wpw[threadIdx.x];
    __syncthreads();

    int n = n0 + lane;
    float y[32];
    #pragma unroll
    for (int c = 0; c < 32; ++c) y[c] = xls[wave][lane][c];

    for (int i = 0; i < 2; ++i) {
        float z[32];
        #pragma unroll
        for (int co = 0; co < 32; ++co) {
            float s = bls[i * 32 + co];
            #pragma unroll
            for (int ci = 0; ci < 32; ++ci) s += y[ci] * wls[(i * 32 + ci) * 32 + co];
            z[co] = (s > 0.0f) ? s : 0.0f;
        }
        #pragma unroll
        for (int c = 0; c < 32; ++c) y[c] = z[c];
    }
    float s = 0.0f;
    #pragma unroll
    for (int c = 0; c < 32; ++c) s += y[c] * wp[c];
    if (n < N_NODES) atomicAdd(&out[batch[n]], s);
}

// ---------------------------------------------------------------------------
// Launch
// ---------------------------------------------------------------------------
extern "C" void kernel_launch(void* const* d_in, const int* in_sizes, int n_in,
                              void* d_out, int out_size, void* d_ws, size_t ws_size,
                              hipStream_t stream) {
    const int*   x_types = (const int*)d_in[0];
    const int*   eidx    = (const int*)d_in[1];
    const float* eattr   = (const float*)d_in[2];
    const int*   batch   = (const int*)d_in[3];
    const float* embd    = (const float*)d_in[4];
    const float* fc1     = (const float*)d_in[5];
    const float* fc2     = (const float*)d_in[6];
    const float* linw    = (const float*)d_in[7];
    const float* linb    = (const float*)d_in[8];
    const float* wpw     = (const float*)d_in[9];
    const float* wpb     = (const float*)d_in[10];
    float* out = (float*)d_out;

    // Workspace carving (~23.5 MB)
    char* ws = (char*)d_ws;
    size_t off = 0;
    auto carve = [&](size_t bytes) { size_t o = off; off += (bytes + 255) & ~(size_t)255; return o; };
    float*     xA   = (float*)(ws + carve((size_t)N_NODES * HDIM * 4));
    float*     xB   = (float*)(ws + carve((size_t)N_NODES * HDIM * 4));
    _Float16*  embf = (_Float16*)(ws + carve((size_t)ET * 32 * 16 * 2));
    _Float16*  fc1f = (_Float16*)(ws + carve((size_t)L_CONV * 16 * 32 * 16 * 2));
    _Float16*  fc2f = (_Float16*)(ws + carve((size_t)L_CONV * 64 * 8 * 32 * 16 * 2));

    // Prep
    k_gather_x<<<(N_NODES * HDIM + 255) / 256, 256, 0, stream>>>(x_types, embd, xA);
    k_fc1_frag<<<(L_CONV * 16 * 32 * 16 + 255) / 256, 256, 0, stream>>>(fc1, fc1f);
    k_fc2_frag<<<(L_CONV * 64 * 8 * 32 * 16 + 255) / 256, 256, 0, stream>>>(fc2, fc2f);
    k_emb_frag<<<(ET * 32 + 255) / 256, 256, 0, stream>>>(eattr, embf);

    // Conv layers (block = 4 waves x 2 tiles = 8 edge tiles)
    float* xin = xA;
    float* xout = xB;
    for (int l = 0; l < L_CONV; ++l) {
        k_zero<<<(N_NODES * HDIM + 255) / 256, 256, 0, stream>>>(xout, N_NODES * HDIM);
        k_conv<<<(ET + 7) / 8, 128, 0, stream>>>(xin, xout, eidx, embf,
                                                 fc1f + (size_t)l * 16 * 32 * 16,
                                                 fc2f + (size_t)l * 64 * 8 * 32 * 16);
        float* tmp = xin; xin = xout; xout = tmp;
    }

    // Head
    k_out_init<<<(N_GRAPH + 255) / 256, 256, 0, stream>>>(wpb, out);
    k_post<<<(N_NODES + 255) / 256, 256, 0, stream>>>(xin, batch, linw, linb, wpw, out);
}
